// Decoder_45483703665104
// MI455X (gfx1250) — matrix operand order repaired and checked by reading, not compiled, not run on hardware
//
#include <hip/hip_runtime.h>
#include <hip/hip_bf16.h>

// ---------------------------------------------------------------------------
// Problem constants: V=50257 E=256 H=256 K=3 T=1024 B=32 S=1024
// ---------------------------------------------------------------------------
#define TT    1024           // T
#define BB    32             // B
#define SS    1024           // S
#define EE    256            // E
#define HH    256            // H
#define C2    512            // 2*H
#define LL    1023           // conv output length
#define M1    (TT*BB)        // 32768
#define M2    (BB*LL)        // 32736
#define KCONV (3*C2)         // 1536

typedef __attribute__((ext_vector_type(16))) __bf16 v16bf;
typedef __attribute__((ext_vector_type(8)))  __bf16 v8bf;
typedef __attribute__((ext_vector_type(8)))  float  v8f;

// ---------------------------------------------------------------------------
// WMMA + fragment-order swizzle helpers (wave32, contiguous-k layouts shared by A and B)
// A 16x32 bf16 : lane = (r&15) | (((k>>4)&1)<<4) ; elem = k&15
// B 32x16 bf16 : lane = (n&15) | (((k>>4)&1)<<4) ; elem = k&15
// Tiles are staged in LDS as v16bf[block][lane] so a fragment load is ONE
// 32-byte aligned vector load per lane (2x ds_load_b128, no packing VALU).
// ---------------------------------------------------------------------------
__device__ __forceinline__ v8f wmma_bf16(v16bf a, v16bf b, v8f c) {
  return __builtin_amdgcn_wmma_f32_16x16x32_bf16(false, a, false, b, (short)0, c,
                                                 false, false);
}

__device__ __forceinline__ v8bf cvt8(const float* s) {
  v8bf v;
#pragma unroll
  for (int u = 0; u < 8; ++u) v[u] = (__bf16)s[u];
  return v;
}
__device__ __forceinline__ v16bf cvt16(const float* s) {
  v16bf v;
#pragma unroll
  for (int u = 0; u < 16; ++u) v[u] = (__bf16)s[u];
  return v;
}
__device__ __forceinline__ v8bf zero8() {
  v8bf v;
#pragma unroll
  for (int u = 0; u < 8; ++u) v[u] = (__bf16)0.0f;
  return v;
}

// dest slot for an 8-run of A elements starting at (r, c8) inside a 32-k chunk
// (c8 multiple of 8): block = mtile*32 + lane ; 8 elems contiguous at eb
__device__ __forceinline__ __bf16* a_dst8(v16bf* sAf, int r, int c8) {
  const int lane = (r & 15) + (((c8 >> 4) & 1) << 4);
  const int eb   = c8 & 15;
  return (__bf16*)&sAf[(r >> 4) * 32 + lane] + eb;
}
// dest slot for a 16-run of B elements at (kb, n), kb in {0,16}
__device__ __forceinline__ v16bf* b_dst16(v16bf* sBf, int kb, int n) {
  return &sBf[(n >> 4) * 32 + (n & 15) + ((kb >> 4) << 4)];
}

// ---------------------------------------------------------------------------
// K1: x = emb[target] @ affine_w.T + affine_b -> bf16 X[b][t][c]
// 128x64 tile, 8 waves 4x2, wave 32x32 (2x2 frags), K=256
// ---------------------------------------------------------------------------
__global__ void __launch_bounds__(256)
k1_embed_affine(const int* __restrict__ target, const float* __restrict__ emb,
                const float* __restrict__ aw, const float* __restrict__ ab,
                __bf16* __restrict__ X) {
  __shared__ int   sId[128];
  __shared__ v16bf sAf[8 * 32];   // 8 m-tiles of 16 rows
  __shared__ v16bf sBf[4 * 32];   // 4 n-tiles of 16 cols
  const int tid = threadIdx.x;
  const int m0 = blockIdx.x * 128, n0 = blockIdx.y * 64;
  if (tid < 128) sId[tid] = target[m0 + tid];
  const int wave = tid >> 5, lane = tid & 31;
  const int wm = (wave >> 1) * 2, wn = (wave & 1) * 2;   // tile indices
  v8f acc[2][2] = {};
  for (int k0 = 0; k0 < EE; k0 += 32) {
    __syncthreads();
    for (int i = tid; i < 512; i += 256) {               // A: 128x32, 8-elem runs
      const int r = i >> 2, c8 = (i & 3) << 3;
      *(v8bf*)a_dst8(sAf, r, c8) = cvt8(&emb[(size_t)sId[r] * EE + k0 + c8]);
    }
    if (tid < 128) {                                     // B: 32x64, 16-elem runs
      const int n = tid & 63, kb = (tid >> 6) << 4;
      *b_dst16(sBf, kb, n) = cvt16(&aw[(size_t)(n0 + n) * EE + k0 + kb]);
    }
    __syncthreads();
#pragma unroll
    for (int mi = 0; mi < 2; ++mi) {
      const v16bf a = sAf[(wm + mi) * 32 + lane];
#pragma unroll
      for (int ni = 0; ni < 2; ++ni)
        acc[mi][ni] = wmma_bf16(a, sBf[(wn + ni) * 32 + lane], acc[mi][ni]);
    }
  }
  const int rbase = 8 * (lane >> 4), col = lane & 15;
#pragma unroll
  for (int mi = 0; mi < 2; ++mi)
#pragma unroll
    for (int ni = 0; ni < 2; ++ni) {
      const int n = n0 + (wn + ni) * 16 + col;
      const float bias = ab[n];
#pragma unroll
      for (int j = 0; j < 8; ++j) {
        const int m = m0 + (wm + mi) * 16 + rbase + j;
        const int t = m >> 5, b = m & 31;                // m = t*32 + b
        X[((size_t)b * TT + t) * C2 + n] = (__bf16)(acc[mi][ni][j] + bias);
      }
    }
}

// ---------------------------------------------------------------------------
// K2: causal conv as GEMM (K=1536 over shifted X rows), +bias, ReLU -> Y f32
// ---------------------------------------------------------------------------
__global__ void __launch_bounds__(256)
k2_conv(const __bf16* __restrict__ X, const float* __restrict__ cw,
        const float* __restrict__ cb, float* __restrict__ Y) {
  __shared__ int   sBi[128], sTi[128];
  __shared__ v16bf sAf[8 * 32];
  __shared__ v16bf sBf[4 * 32];
  const int tid = threadIdx.x;
  const int m0 = blockIdx.x * 128, n0 = blockIdx.y * 64;
  if (tid < 128) {
    const int m = m0 + tid;
    sBi[tid] = (m < M2) ? (m / LL) : 0;
    sTi[tid] = (m < M2) ? (m % LL) : -1000;
  }
  const int wave = tid >> 5, lane = tid & 31;
  const int wm = (wave >> 1) * 2, wn = (wave & 1) * 2;
  v8f acc[2][2] = {};
  for (int k0 = 0; k0 < KCONV; k0 += 32) {
    __syncthreads();
    const int dk = k0 >> 9;            // 512 % 32 == 0 -> constant per chunk
    const int cbase = k0 & 511;
    for (int i = tid; i < 512; i += 256) {   // A: bf16 copy of shifted X rows
      const int r = i >> 2, c8 = (i & 3) << 3;
      const int tr = sTi[r] - 2 + dk;
      v8bf v = zero8();
      if (tr >= 0)
        v = *(const v8bf*)&X[((size_t)sBi[r] * TT + tr) * C2 + cbase + c8];
      *(v8bf*)a_dst8(sAf, r, c8) = v;
    }
    if (tid < 128) {                         // B[k][n] = conv_w[n][0][dk][c]
      const int n = tid & 63, kb = (tid >> 6) << 4;
      *b_dst16(sBf, kb, n) = cvt16(&cw[(size_t)(n0 + n) * KCONV + k0 + kb]);
    }
    __syncthreads();
#pragma unroll
    for (int mi = 0; mi < 2; ++mi) {
      const v16bf a = sAf[(wm + mi) * 32 + lane];
#pragma unroll
      for (int ni = 0; ni < 2; ++ni)
        acc[mi][ni] = wmma_bf16(a, sBf[(wn + ni) * 32 + lane], acc[mi][ni]);
    }
  }
  const int rbase = 8 * (lane >> 4), col = lane & 15;
#pragma unroll
  for (int mi = 0; mi < 2; ++mi)
#pragma unroll
    for (int ni = 0; ni < 2; ++ni) {
      const int n = n0 + (wn + ni) * 16 + col;
      const float bias = cb[n];
#pragma unroll
      for (int j = 0; j < 8; ++j) {
        const int m = m0 + (wm + mi) * 16 + rbase + j;
        if (m < M2) Y[(size_t)m * C2 + n] = fmaxf(acc[mi][ni][j] + bias, 0.0f);
      }
    }
}

// ---------------------------------------------------------------------------
// K3: dec = Y[:, :256] * softmax(Y[:, 256:512]) ; one wave per row -> bf16
// ---------------------------------------------------------------------------
__global__ void __launch_bounds__(256)
k3_glu(const float* __restrict__ Y, __bf16* __restrict__ D) {
  const int wave = threadIdx.x >> 5, lane = threadIdx.x & 31;
  const int m = blockIdx.x * 8 + wave;
  if (m >= M2) return;
  const float* row = Y + (size_t)m * C2;
  float g[8];
  float mx = -3.0e38f;
#pragma unroll
  for (int i = 0; i < 8; ++i) { g[i] = row[HH + lane * 8 + i]; mx = fmaxf(mx, g[i]); }
#pragma unroll
  for (int d = 16; d >= 1; d >>= 1) mx = fmaxf(mx, __shfl_xor(mx, d, 32));
  float s = 0.0f;
#pragma unroll
  for (int i = 0; i < 8; ++i) { g[i] = __expf(g[i] - mx); s += g[i]; }
#pragma unroll
  for (int d = 16; d >= 1; d >>= 1) s += __shfl_xor(s, d, 32);
  const float inv = 1.0f / s;
#pragma unroll
  for (int i = 0; i < 8; ++i)
    D[(size_t)m * HH + lane * 8 + i] = (__bf16)(row[lane * 8 + i] * g[i] * inv);
}

// ---------------------------------------------------------------------------
// K4: dec_attn2 = D @ map_w.T + map_b -> d_out (initializes every element)
// ---------------------------------------------------------------------------
__global__ void __launch_bounds__(256)
k4_map(const __bf16* __restrict__ D, const float* __restrict__ mw,
       const float* __restrict__ mb, float* __restrict__ out) {
  __shared__ v16bf sAf[8 * 32];
  __shared__ v16bf sBf[4 * 32];
  const int tid = threadIdx.x;
  const int m0 = blockIdx.x * 128, n0 = blockIdx.y * 64;
  const int wave = tid >> 5, lane = tid & 31;
  const int wm = (wave >> 1) * 2, wn = (wave & 1) * 2;
  v8f acc[2][2] = {};
  for (int k0 = 0; k0 < HH; k0 += 32) {
    __syncthreads();
    for (int i = tid; i < 512; i += 256) {
      const int r = i >> 2, c8 = (i & 3) << 3;
      const int m = m0 + r;
      v8bf v = zero8();
      if (m < M2) v = *(const v8bf*)&D[(size_t)m * HH + k0 + c8];
      *(v8bf*)a_dst8(sAf, r, c8) = v;
    }
    if (tid < 128) {                         // B[k][n] = map_w[n][k]
      const int n = tid & 63, kb = (tid >> 6) << 4;
      *b_dst16(sBf, kb, n) = cvt16(&mw[(size_t)(n0 + n) * HH + k0 + kb]);
    }
    __syncthreads();
#pragma unroll
    for (int mi = 0; mi < 2; ++mi) {
      const v16bf a = sAf[(wm + mi) * 32 + lane];
#pragma unroll
      for (int ni = 0; ni < 2; ++ni)
        acc[mi][ni] = wmma_bf16(a, sBf[(wn + ni) * 32 + lane], acc[mi][ni]);
    }
  }
  const int rbase = 8 * (lane >> 4), col = lane & 15;
#pragma unroll
  for (int mi = 0; mi < 2; ++mi)
#pragma unroll
    for (int ni = 0; ni < 2; ++ni) {
      const int n = n0 + (wn + ni) * 16 + col;
      const float bias = mb[n];
#pragma unroll
      for (int j = 0; j < 8; ++j) {
        const int m = m0 + (wm + mi) * 16 + rbase + j;
        if (m < M2) out[(size_t)m * C2 + n] = acc[mi][ni][j] + bias;
      }
    }
}

// ---------------------------------------------------------------------------
// K5: attention. WG = (batch b, 32 queries). Scores 32x1024 f32 in LDS,
// softmax, P @ V, += into d_out. All operands staged in fragment order.
// LDS carve (all 32B aligned):
//   sS   f32 [32][1024]                 131072 B
//   sPf  v16bf [32 kc][2 mt][32 lane]    65536 B
//   sQf  v16bf [ 8 kc][2 mt][32 lane]    16384 B
//   sKVf v16bf [32 nb][32 lane]          32768 B
// ---------------------------------------------------------------------------
#define K5_SMEM (131072 + 65536 + 16384 + 32768)

__global__ void __launch_bounds__(256)
k5_attn(const __bf16* __restrict__ D, const float* __restrict__ enc,
        const float* __restrict__ vsrc, float* __restrict__ out) {
  extern __shared__ char smem[];
  float* sS   = (float*)smem;
  v16bf* sPf  = (v16bf*)(smem + 131072);
  v16bf* sQf  = (v16bf*)(smem + 131072 + 65536);
  v16bf* sKVf = (v16bf*)(smem + 131072 + 65536 + 16384);
  const int tid = threadIdx.x;
  const int b = blockIdx.x, l0 = blockIdx.y * 32;
  const int wave = tid >> 5, lane = tid & 31;

  // stage Q (32 x 256) in A-fragment order, zero-pad rows >= LL
  for (int i = tid; i < 32 * 32; i += 256) {     // 8-elem runs
    const int r = i >> 5, cc = (i & 31) << 3;    // cc in [0,256)
    const int l = l0 + r;
    v8bf v = zero8();
    if (l < LL) v = *(const v8bf*)&D[((size_t)b * LL + l) * HH + cc];
    const int kc = cc >> 5, c8 = cc & 31;
    *(v8bf*)a_dst8(&sQf[kc * 2 * 32], r, c8) = v;
  }

  // phase 1: scores = Q @ K^T (two 512-col passes; wave owns 64 cols = 4 frags)
  for (int nb = 0; nb < 2; ++nb) {
    v8f acc[2][4] = {};
    for (int k0 = 0; k0 < HH; k0 += 32) {
      __syncthreads();
      if (tid < 128) {                           // not enough 16-runs for all:
        // 32x512 tile, 16-elem k-runs: 512 n * 2 kb = 1024 chunks, 4 per thread
      }
      for (int i = tid; i < 1024; i += 256) {    // Kt[k][s]=enc[b][s][k], k-contig
        const int n = i & 511, kb = (i >> 9) << 4;
        *b_dst16(sKVf, kb, n) =
            cvt16(&enc[((size_t)b * SS + nb * 512 + n) * HH + k0 + kb]);
      }
      __syncthreads();
#pragma unroll
      for (int mi = 0; mi < 2; ++mi) {
        const v16bf a = sQf[((k0 >> 5) * 2 + mi) * 32 + lane];
#pragma unroll
        for (int ni = 0; ni < 4; ++ni)
          acc[mi][ni] = wmma_bf16(a, sKVf[(wave * 4 + ni) * 32 + lane], acc[mi][ni]);
      }
    }
    const int rbase = 8 * (lane >> 4), col = lane & 15;
#pragma unroll
    for (int mi = 0; mi < 2; ++mi)
#pragma unroll
      for (int ni = 0; ni < 4; ++ni)
#pragma unroll
        for (int j = 0; j < 8; ++j)
          sS[(mi * 16 + rbase + j) * 1024 + nb * 512 + wave * 64 + ni * 16 + col] =
              acc[mi][ni][j];
  }
  __syncthreads();

  // phase 2: row softmax (wave per row) -> P in A-fragment order (bf16)
  for (int r = wave; r < 32; r += 8) {
    float mx = -3.0e38f;
    for (int c = lane; c < 1024; c += 32) mx = fmaxf(mx, sS[r * 1024 + c]);
#pragma unroll
    for (int d = 16; d >= 1; d >>= 1) mx = fmaxf(mx, __shfl_xor(mx, d, 32));
    float sum = 0.0f;
    for (int c = lane; c < 1024; c += 32) sum += __expf(sS[r * 1024 + c] - mx);
#pragma unroll
    for (int d = 16; d >= 1; d >>= 1) sum += __shfl_xor(sum, d, 32);
    const float inv = 1.0f / sum;
    for (int c = lane; c < 1024; c += 32) {
      const float p = __expf(sS[r * 1024 + c] - mx) * inv;
      const int kc = c >> 5, cc = c & 31;
      const int ln = (r & 15) + (((cc >> 4) & 1) << 4);
      const int el = cc & 15;
      ((__bf16*)&sPf[(kc * 2 + (r >> 4)) * 32 + ln])[el] = (__bf16)p;
    }
  }

  // phase 3: O = P @ V ; += into out
  v8f oacc[2][4] = {};
  for (int k0 = 0; k0 < SS; k0 += 32) {
    __syncthreads();
    for (int i = tid; i < 32 * 128; i += 256) {  // V[s][d] f32, n-contig float4
      const int n4 = (i & 127) << 2, k = i >> 7;
      const float* s = &vsrc[((size_t)b * SS + k0 + k) * C2 + n4];
#pragma unroll
      for (int u = 0; u < 4; ++u) {
        const int n = n4 + u;
        ((__bf16*)&sKVf[(n >> 4) * 32 + (n & 15) + (((k >> 4) & 1) << 4)])[k & 15] =
            (__bf16)s[u];
      }
    }
    __syncthreads();
#pragma unroll
    for (int mi = 0; mi < 2; ++mi) {
      const v16bf a = sPf[((k0 >> 5) * 2 + mi) * 32 + lane];
#pragma unroll
      for (int ni = 0; ni < 4; ++ni)
        oacc[mi][ni] = wmma_bf16(a, sKVf[(wave * 4 + ni) * 32 + lane], oacc[mi][ni]);
    }
  }
  const int rbase = 8 * (lane >> 4), col = lane & 15;
#pragma unroll
  for (int mi = 0; mi < 2; ++mi)
#pragma unroll
    for (int ni = 0; ni < 4; ++ni) {
      const int d = wave * 64 + ni * 16 + col;
#pragma unroll
      for (int j = 0; j < 8; ++j) {
        const int l = l0 + mi * 16 + rbase + j;
        if (l < LL) out[((size_t)b * LL + l) * C2 + d] += oacc[mi][ni][j];
      }
    }
}

// ---------------------------------------------------------------------------
// launch. inputs: 0 source(unused) 1 target 2 enc_attn 3 source_seq_out 4 emb
// 5 affine_w 6 affine_b 7 conv_w 8 conv_b 9 map_w 10 map_b
// ws carve: X bf16 32M | Y f32 64M | D bf16 16M  (~117 MB)
// ---------------------------------------------------------------------------
extern "C" void kernel_launch(void* const* d_in, const int* in_sizes, int n_in,
                              void* d_out, int out_size, void* d_ws, size_t ws_size,
                              hipStream_t stream) {
  (void)in_sizes; (void)n_in; (void)out_size; (void)ws_size;
  const int*   target = (const int*)d_in[1];
  const float* enc    = (const float*)d_in[2];
  const float* vsrc   = (const float*)d_in[3];
  const float* emb    = (const float*)d_in[4];
  const float* aw     = (const float*)d_in[5];
  const float* ab     = (const float*)d_in[6];
  const float* cw     = (const float*)d_in[7];
  const float* cb     = (const float*)d_in[8];
  const float* mw     = (const float*)d_in[9];
  const float* mb     = (const float*)d_in[10];
  float* out = (float*)d_out;

  char* ws = (char*)d_ws;
  __bf16* X = (__bf16*)ws;                            // 33,554,432 B
  float*  Y = (float*)(ws + 33554432);                // 67,043,328 B
  __bf16* D = (__bf16*)(ws + 33554432 + 67043328);    // 16,760,832 B

  k1_embed_affine<<<dim3(M1 / 128, C2 / 64), 256, 0, stream>>>(target, emb, aw, ab, X);
  k2_conv<<<dim3((M2 + 127) / 128, C2 / 64), 256, 0, stream>>>(X, cw, cb, Y);
  k3_glu<<<dim3((M2 + 7) / 8), 256, 0, stream>>>(Y, D);
  k4_map<<<dim3((M2 + 127) / 128, C2 / 64), 256, 0, stream>>>(D, mw, mb, out);
  k5_attn<<<dim3(BB, (LL + 31) / 32), 256, K5_SMEM, stream>>>(D, enc, vsrc, out);
}
